// EnergyBasedEventGenerator_34162169872774
// MI455X (gfx1250) — matrix operand order repaired and checked by reading, not compiled, not run on hardware
//
#include <hip/hip_runtime.h>

// ---------------------------------------------------------------------------
// EnergyBasedEventGenerator on MI455X (gfx1250, wave32, WMMA)
//
// Everything after the mass-spring recurrence is LINEAR in the per-step force
// vector (to_samples -> complex filter -> irfft -> Hann window), so it fuses
// into one precomputed C[128,512] matrix. Additionally, overlap-add consumes
// every GEMM output element exactly once, so OA fuses into the GEMM epilogue:
// each wave computes one 16x16 OUTPUT tile as two chained 4-WMMA passes
// (frame t0 cols n..n+15, frame t0-1 cols n+256..) into a single accumulator.
// Same 524288 WMMAs as the unfused GEMM, but the 256MB blocks round-trip is
// gone: traffic ~144MB (forces 32R + fseq 16W/32R + out 64W) -> ~6us @23.3TB/s.
// B fragments (8KB, shared by all 8 waves of a block) are staged in LDS via
// the gfx1250 async-to-LDS path when available.
// ---------------------------------------------------------------------------

#define BLOCK      512
#define N_COEFFS   257            // BLOCK/2 + 1
#define N_FRAMES   512
#define N_SAMPLES  131072         // 2^17
#define Bdim       8
#define Edim       16
#define Ddim       128
#define BE         (Bdim*Edim)    // 128
#define BED        (BE*Ddim)      // 16384
#define PI_F       3.14159265358979323846f

typedef __attribute__((ext_vector_type(16))) __bf16 v16bf;
typedef __attribute__((ext_vector_type(8)))  __bf16 v8bf;
typedef __attribute__((ext_vector_type(8)))  float  v8f;
typedef __attribute__((ext_vector_type(4)))  int    v4i;

#if defined(__has_builtin)
#if __has_builtin(__builtin_amdgcn_global_load_async_to_lds_b128) && \
    __has_builtin(__builtin_amdgcn_s_wait_asynccnt)
#define HAVE_ASYNC_LDS 1
#endif
#endif

#ifdef HAVE_ASYNC_LDS
typedef __attribute__((address_space(1))) v4i* g4i_t;   // global int4*
typedef __attribute__((address_space(3))) v4i* l4i_t;   // LDS int4*
__device__ __forceinline__ void async_copy16(const void* g, void* l) {
    // generic->AS1 and generic->AS3 via integer round-trip (low 32 bits of a
    // generic LDS address are the LDS offset on AMDGPU)
    __builtin_amdgcn_global_load_async_to_lds_b128(
        (g4i_t)(unsigned long long)g,
        (l4i_t)(unsigned int)(unsigned long long)l, 0, 0);
}
#endif

__device__ __forceinline__ unsigned short f2bf(float f) {
    unsigned int x = __float_as_uint(f);
    unsigned int r = x + 0x7FFFu + ((x >> 16) & 1u);   // round-to-nearest-even
    return (unsigned short)(r >> 16);
}

// ---------------------------------------------------------------------------
// K0a: G[d][f][2] = sum_k ( S[2k,d], S[2k+1,d] ) * complex F[k,f]
// ---------------------------------------------------------------------------
__global__ __launch_bounds__(256) void k_compose_filter(
    const float* __restrict__ S,      // [514,128] to_samples
    const float* __restrict__ F,      // [257,257,2] filters (re,im)
    float* __restrict__ G)            // [128,257,2]
{
    int tid = blockIdx.x * blockDim.x + threadIdx.x;
    if (tid >= Ddim * N_COEFFS) return;
    int d = tid & (Ddim - 1);
    int f = tid >> 7;
    float gre = 0.f, gim = 0.f;
    for (int k = 0; k < N_COEFFS; ++k) {
        float sre = S[(2 * k) * Ddim + d];
        float sim = S[(2 * k + 1) * Ddim + d];
        float fre = F[(k * N_COEFFS + f) * 2 + 0];
        float fim = F[(k * N_COEFFS + f) * 2 + 1];
        gre += sre * fre - sim * fim;
        gim += sre * fim + sim * fre;
    }
    G[(d * N_COEFFS + f) * 2 + 0] = gre;
    G[(d * N_COEFFS + f) * 2 + 1] = gim;
}

// ---------------------------------------------------------------------------
// K0b: fold irfft basis + Hann window into C[k=d][n], emitted in WMMA
//      B-fragment order (bf16): Bfrag[((ntile*4+kchunk)*32+lane)*16 + e],
//      col n = ntile*16+(lane&15), k = kchunk*32 + (lane>>4)*16 + e
// ---------------------------------------------------------------------------
__global__ __launch_bounds__(256) void k_build_bfrag(
    const float* __restrict__ G,          // [128,257,2]
    unsigned short* __restrict__ Bfrag)   // 32*4*32*16 bf16 (128KB)
{
    int u = blockIdx.x * blockDim.x + threadIdx.x;   // 65536 threads
    int e      = u & 15;
    int lane   = (u >> 4) & 31;
    int kchunk = (u >> 9) & 3;
    int ntile  = u >> 11;
    int n = ntile * 16 + (lane & 15);
    int k = kchunk * 32 + (lane >> 4) * 16 + e;      // = d index into G

    float phi  = 2.f * PI_F * (float)n / (float)BLOCK;
    float win  = 0.5f - 0.5f * cosf(phi);
    float scal = win * (1.f / (float)BLOCK);

    float cphi = cosf(phi), sphi = sinf(phi);
    float c = 1.f, s = 0.f;                           // cos/sin(f*phi), f=0
    float acc = 0.f;
    const float* g = G + (size_t)k * N_COEFFS * 2;
    for (int f = 0; f < N_COEFFS; ++f) {
        float sc  = (f == 0 || f == N_COEFFS - 1) ? 1.f : 2.f;
        float gre = g[2 * f], gim = g[2 * f + 1];
        acc += sc * (gre * c - gim * s);
        float nc = c * cphi - s * sphi;               // rotate angle by phi
        s = s * cphi + c * sphi;
        c = nc;
    }
    Bfrag[u] = f2bf(acc * scal);
}

// ---------------------------------------------------------------------------
// K1: mass-spring recurrence. One thread per (b,e,d); force = f - state*t.
//     Writes bf16 force_seq[t][be*128+d] == row-major WMMA A operand
//     [r = t*128+be][k = d]; per-t stores are lane-contiguous (coalesced).
// ---------------------------------------------------------------------------
__global__ __launch_bounds__(256) void k_recurrence(
    const float* __restrict__ masses,
    const float* __restrict__ tensions,
    const float* __restrict__ damping,
    const float* __restrict__ forces,         // [BED, 512] (t contiguous)
    unsigned short* __restrict__ fseq)        // [512, BED] bf16
{
    int bed = blockIdx.x * blockDim.x + threadIdx.x;  // 16384 threads
    float m   = 1.f + fabsf(masses[bed]) * 100.f;
    float tt  = 0.01f + fabsf(tensions[bed]) * 10.f;
    float sig = 1.f / (1.f + __expf(-damping[bed]));
    float dd  = 0.5f + sig * 0.5f * 0.9999f;          // BASE_DAMP + sigmoid*SPAN*0.9999
    float inv_m = 1.f / m;

    float state = 0.f, vel = 0.f;
    const float4* frow = reinterpret_cast<const float4*>(forces + (size_t)bed * N_FRAMES);
    for (int t4 = 0; t4 < N_FRAMES / 4; ++t4) {
        float4 f4 = frow[t4];
        float fv[4] = { f4.x, f4.y, f4.z, f4.w };
        #pragma unroll
        for (int j = 0; j < 4; ++j) {
            float force = fv[j] - state * tt;         // == m*acc of the reference
            float acc   = force * inv_m;
            vel  += acc;
            state += vel;
            vel  *= dd;
            fseq[(size_t)(t4 * 4 + j) * BED + bed] = f2bf(force);
        }
    }
}

// ---------------------------------------------------------------------------
// K2: fused WMMA GEMM + overlap-add. Block = one 16-sample s-tile (uniform
//     t0/nt0), 8 waves = 8 be-tiles. Wave computes the 16x16 OUTPUT tile:
//       out = A(t0 rows)   x B(cols nt0)      [4 WMMAs]
//           + A(t0-1 rows) x B(cols nt0+16)   [4 WMMAs, skipped when t0==0]
//     B fragments (2 x 4KB contiguous) staged in LDS, shared by all waves.
// ---------------------------------------------------------------------------
__global__ __launch_bounds__(256) void k_gemm_oa(
    const __bf16* __restrict__ A,        // [65536,128] bf16; row r = t*128+be
    const __bf16* __restrict__ Bf,       // B fragments (see k_build_bfrag)
    float* __restrict__ out)             // [128, 131072] flat
{
    __shared__ __align__(32) unsigned char smem[8192];
    const int st   = blockIdx.x;          // 8192 s-tiles
    const int t0   = st >> 4;             // frame covering these samples
    const int nt0  = st & 15;             // B column tile within frame t0
    const int s0   = st << 4;
    const int lane = threadIdx.x & 31;
    const int bt   = threadIdx.x >> 5;    // be-tile = wave id in block
    const int khalf = lane >> 4;

    // ---- stage the two 4KB B-fragment spans into LDS (async if available)
    {
        const unsigned char* gb = (const unsigned char*)Bf;
        const unsigned int off = threadIdx.x * 16;      // 256 thr x 16B = 4KB
#ifdef HAVE_ASYNC_LDS
        async_copy16(gb + (size_t)nt0 * 4096 + off,        smem + off);
        async_copy16(gb + (size_t)(nt0 + 16) * 4096 + off, smem + 4096 + off);
        __builtin_amdgcn_s_wait_asynccnt(0);
#else
        *(uint4*)(smem + off)        = *(const uint4*)(gb + (size_t)nt0 * 4096 + off);
        *(uint4*)(smem + 4096 + off) = *(const uint4*)(gb + (size_t)(nt0 + 16) * 4096 + off);
#endif
    }
    __syncthreads();

    // ---- A fragments (16x32 bf16, contiguous k = 16*(lane>>4) + e): frame t0 and t0-1
    v16bf a0[4], a1[4];
    const int r0 = t0 * BE + bt * 16 + (lane & 15);
    #pragma unroll
    for (int kc = 0; kc < 4; ++kc) {
        const v8bf* p = reinterpret_cast<const v8bf*>(
            A + (size_t)r0 * Ddim + kc * 32 + khalf * 16);
        v8bf lo = p[0], hi = p[1];
        #pragma unroll
        for (int i = 0; i < 8; ++i) { a0[kc][i] = lo[i]; a0[kc][i + 8] = hi[i]; }
    }
    if (t0 > 0) {                                  // wave-uniform branch
        #pragma unroll
        for (int kc = 0; kc < 4; ++kc) {
            const v8bf* p = reinterpret_cast<const v8bf*>(
                A + (size_t)(r0 - BE) * Ddim + kc * 32 + khalf * 16);
            v8bf lo = p[0], hi = p[1];
            #pragma unroll
            for (int i = 0; i < 8; ++i) { a1[kc][i] = lo[i]; a1[kc][i + 8] = hi[i]; }
        }
    }

    // ---- chained WMMAs: both frames accumulate into one 16x16 f32 tile
    v8f c = {0.f, 0.f, 0.f, 0.f, 0.f, 0.f, 0.f, 0.f};
    #pragma unroll
    for (int kc = 0; kc < 4; ++kc) {
        v16bf b = *reinterpret_cast<const v16bf*>(smem + kc * 1024 + lane * 32);
        c = __builtin_amdgcn_wmma_f32_16x16x32_bf16(
                false, a0[kc], false, b, (short)0, c, false, false);
    }
    if (t0 > 0) {                                  // wave-uniform, EXEC stays all-1
        #pragma unroll
        for (int kc = 0; kc < 4; ++kc) {
            v16bf b = *reinterpret_cast<const v16bf*>(smem + 4096 + kc * 1024 + lane * 32);
            c = __builtin_amdgcn_wmma_f32_16x16x32_bf16(
                    false, a1[kc], false, b, (short)0, c, false, false);
        }
    }

    // ---- store: D layout M = vgpr + 8*(lane>>4), N = lane&15
    float* op = out + (size_t)(bt * 16 + 8 * khalf) * N_SAMPLES + s0 + (lane & 15);
    #pragma unroll
    for (int v = 0; v < 8; ++v)
        op[(size_t)v * N_SAMPLES] = c[v];
}

// ---------------------------------------------------------------------------
extern "C" void kernel_launch(void* const* d_in, const int* in_sizes, int n_in,
                              void* d_out, int out_size, void* d_ws, size_t ws_size,
                              hipStream_t stream) {
    const float* masses     = (const float*)d_in[0];
    const float* tensions   = (const float*)d_in[1];
    const float* damping    = (const float*)d_in[2];
    const float* forces     = (const float*)d_in[3];
    const float* to_samples = (const float*)d_in[4];
    const float* filters    = (const float*)d_in[5];
    float* out = (float*)d_out;

    char* ws = (char*)d_ws;
    // workspace layout (bytes), total 17,171,456
    float*          G     = (float*)(ws);                    // 263,168
    unsigned short* Bfrag = (unsigned short*)(ws + 263168);  // 131,072
    unsigned short* Fseq  = (unsigned short*)(ws + 394240);  // 16,777,216

    // K0a: to_samples x complex-filters composition
    k_compose_filter<<<(Ddim * N_COEFFS + 255) / 256, 256, 0, stream>>>(
        to_samples, filters, G);
    // K0b: fold irfft + window, emit bf16 WMMA B fragments
    k_build_bfrag<<<65536 / 256, 256, 0, stream>>>(G, Bfrag);
    // K1: sequential recurrence -> bf16 A operand
    k_recurrence<<<BED / 256, 256, 0, stream>>>(
        masses, tensions, damping, forces, Fseq);
    // K2: fused GEMM + overlap-add (one wave per 16x16 output tile)
    k_gemm_oa<<<N_SAMPLES / 16, 256, 0, stream>>>(
        (const __bf16*)Fseq, (const __bf16*)Bfrag, out);
}
